// GraphClassifier_747324309628
// MI455X (gfx1250) — matrix operand order repaired and checked by reading, not compiled, not run on hardware
//
#include <hip/hip_runtime.h>
#include <hip/hip_bf16.h>

typedef __attribute__((ext_vector_type(16))) _Float16 v16h;
typedef __attribute__((ext_vector_type(8)))  float    v8f;

static constexpr int   NN  = 50000;      // nodes
static constexpr int   EE  = 800000;     // edges (without self loops)
static constexpr int   FD  = 128;        // feature dim (F == H)
static constexpr int   GG  = 10;         // graphs
static constexpr int   CC  = 10;         // classes
static constexpr int   ET  = EE + NN;    // edges incl. self loops
static constexpr float NEG = 0.2f;

// ---------------------------------------------------------------- utilities
__global__ void fill_f32_kernel(float* p, float v, int n) {
  int i = blockIdx.x * blockDim.x + threadIdx.x;
  if (i < n) p[i] = v;
}
__global__ void fill_u32_kernel(unsigned* p, unsigned v, int n) {
  int i = blockIdx.x * blockDim.x + threadIdx.x;
  if (i < n) p[i] = v;
}

// native device-scope fp32 atomic add (no return -> STOREcnt, no CAS loop)
__device__ __forceinline__ void atomic_add_f32_dev(float* p, float v) {
  asm volatile("global_atomic_add_f32 %0, %1, off scope:SCOPE_DEV"
               : : "v"(p), "v"(v) : "memory");
}

// monotonic float<->uint mapping for atomic max on floats
__device__ __forceinline__ unsigned f2ord(float f) {
  unsigned u = __float_as_uint(f);
  return (u & 0x80000000u) ? ~u : (u | 0x80000000u);
}
__device__ __forceinline__ float ord2f(unsigned u) {
  return (u & 0x80000000u) ? __uint_as_float(u & 0x7FFFFFFFu) : __uint_as_float(~u);
}
static constexpr unsigned ORD_NEG_INF = 0x007FFFFFu; // f2ord(-inf)

// W [K=128][N=128] fp32 row-major  ->  Wt [N][K] f16 (column of W contiguous)
__global__ void wtrans_f16_kernel(const float* __restrict__ W, _Float16* __restrict__ Wt) {
  int i = blockIdx.x * blockDim.x + threadIdx.x;   // FD*FD threads
  int n = i >> 7, k = i & (FD - 1);
  Wt[n * FD + k] = (_Float16)W[k * FD + n];
}

// ---------------------------------------------------------------- WMMA GEMM
// C[M x 128] = A[M x 128] * B[128 x 128], B given as f16 transposed (Bt[n][k]).
// One wave computes one 16x16 tile.  MODE: 0 = store, 1 = +bias, 2 = relu(+bias)
template <int MODE>
__global__ void gemm_wmma_kernel(const float* __restrict__ A,
                                 const _Float16* __restrict__ Bt,
                                 const float* __restrict__ bias,
                                 float* __restrict__ C, int M) {
  const int wave = blockIdx.x * (blockDim.x >> 5) + (threadIdx.x >> 5);
  const int lane = threadIdx.x & 31;
  const int tiles_n = FD / 16;                 // 8
  const int tm = wave / tiles_n, tn = wave % tiles_n;
  const int m0 = tm * 16, n0 = tn * 16;
  if (m0 >= M) return;

  const int arow      = m0 + (lane & 15);
  const int acol_base = (lane >> 4) << 4;      // 0 or 16  (A: K 0-15 vs 16-31)
  const int bn        = n0 + (lane & 15);
  const int bk_base   = (lane >> 4) << 4;      // 0 or 16  (B: K 0-15 vs 16-31)

  v8f acc = {};
#pragma unroll
  for (int k0 = 0; k0 < FD; k0 += 32) {
    v16h a, b;
    const float* ap = A + (size_t)arow * FD + k0 + acol_base;
#pragma unroll
    for (int i = 0; i < 8; ++i) a[i] = (_Float16)ap[i];
#pragma unroll
    for (int i = 0; i < 8; ++i) a[8 + i] = (_Float16)ap[8 + i];
    const _Float16* bp = Bt + (size_t)bn * FD + k0 + bk_base;
#pragma unroll
    for (int i = 0; i < 16; ++i) b[i] = bp[i];
    acc = __builtin_amdgcn_wmma_f32_16x16x32_f16(false, a, false, b,
                                                 (short)0, acc, false, false);
  }

  const int crow0 = m0 + ((lane >> 4) << 3);   // rows r (lanes 0-15) / r+8 (lanes 16-31)
  const int ccol  = n0 + (lane & 15);
#pragma unroll
  for (int r = 0; r < 8; ++r) {
    float v = acc[r];
    if (MODE >= 1) v += bias[ccol];
    if (MODE == 2) v = fmaxf(v, 0.0f);
    C[(size_t)(crow0 + r) * FD + ccol] = v;
  }
}

// ---------------------------------------------------------------- GAT pieces
// per-node attention scalars: hs = h.a_src, hd = h.a_dst  (one wave per node)
__global__ void node_scores_kernel(const float* __restrict__ h,
                                   const float* __restrict__ a_src,
                                   const float* __restrict__ a_dst,
                                   float* __restrict__ hs, float* __restrict__ hd) {
  int node = blockIdx.x * (blockDim.x >> 5) + (threadIdx.x >> 5);
  int lane = threadIdx.x & 31;
  if (node >= NN) return;
  const float* row = h + (size_t)node * FD;
  float s1 = 0.f, s2 = 0.f;
#pragma unroll
  for (int k = lane; k < FD; k += 32) {
    float v = row[k];
    s1 += v * a_src[k];
    s2 += v * a_dst[k];
  }
#pragma unroll
  for (int off = 16; off > 0; off >>= 1) {
    s1 += __shfl_xor(s1, off, 32);
    s2 += __shfl_xor(s2, off, 32);
  }
  if (lane == 0) { hs[node] = s1; hd[node] = s2; }
}

__device__ __forceinline__ void edge_sd(const int* __restrict__ ei, int e, int& s, int& d) {
  if (e < EE) { s = ei[e]; d = ei[EE + e]; }   // edge_index rows 0/1
  else        { s = d = e - EE; }              // self loop
}

// pass 1: e = leaky_relu(hs[src]+hd[dst]); segment max over dst (atomic on ordered uint)
__global__ void edge_max_kernel(const int* __restrict__ ei,
                                const float* __restrict__ hs, const float* __restrict__ hd,
                                float* __restrict__ ew, unsigned* __restrict__ mord) {
  int e = blockIdx.x * blockDim.x + threadIdx.x;
  if (e >= ET) return;
  int s, d; edge_sd(ei, e, s, d);
  float v = hs[s] + hd[d];
  v = (v > 0.f) ? v : NEG * v;
  ew[e] = v;
  atomicMax(&mord[d], f2ord(v));
}

// pass 2: w = exp(e - m[dst]); segment sum over dst
__global__ void edge_exp_kernel(const int* __restrict__ ei,
                                float* __restrict__ ew,
                                const unsigned* __restrict__ mord,
                                float* __restrict__ ssum) {
  int e = blockIdx.x * blockDim.x + threadIdx.x;
  if (e >= ET) return;
  int s, d; edge_sd(ei, e, s, d);
  (void)s;
  float w = __expf(ew[e] - ord2f(mord[d]));
  ew[e] = w;
  atomic_add_f32_dev(&ssum[d], w);
}

// pass 3: out[dst] += (w/s[dst]) * h[src]   (one wave per edge, 4 feats per lane)
__global__ void edge_agg_kernel(const int* __restrict__ ei,
                                const float* __restrict__ ew,
                                const float* __restrict__ ssum,
                                const float* __restrict__ h,
                                float* __restrict__ out) {
  int e = blockIdx.x * (blockDim.x >> 5) + (threadIdx.x >> 5);
  if (e >= ET) return;
  int lane = threadIdx.x & 31;
  int s, d; edge_sd(ei, e, s, d);
  float alpha = 0.0f;
  if (lane == 0) alpha = ew[e] / (ssum[d] + 1e-16f);
  alpha = __shfl(alpha, 0, 32);                 // broadcast from lane 0
  const float4 hv = *(const float4*)(h + (size_t)s * FD + lane * 4);
  float* op = out + (size_t)d * FD + lane * 4;
  atomic_add_f32_dev(op + 0, alpha * hv.x);
  atomic_add_f32_dev(op + 1, alpha * hv.y);
  atomic_add_f32_dev(op + 2, alpha * hv.z);
  atomic_add_f32_dev(op + 3, alpha * hv.w);
}

// out = relu(out + b)  elementwise over N x 128
__global__ void bias_relu_kernel(float* __restrict__ h, const float* __restrict__ b) {
  int i = blockIdx.x * blockDim.x + threadIdx.x;
  if (i >= NN * FD) return;
  h[i] = fmaxf(h[i] + b[i & (FD - 1)], 0.0f);
}

// ---------------------------------------------------------------- pooling / head
__global__ void pool_sum_kernel(const float* __restrict__ nf, const int* __restrict__ batch,
                                float* __restrict__ sums, float* __restrict__ cnt) {
  int i = blockIdx.x * blockDim.x + threadIdx.x;
  if (i >= NN * FD) return;
  int node = i >> 7, k = i & (FD - 1);
  int g = batch[node];
  atomic_add_f32_dev(&sums[g * FD + k], nf[i]);
  if (k == 0) atomic_add_f32_dev(&cnt[g], 1.0f);
}
__global__ void pool_div_kernel(const float* __restrict__ sums, const float* __restrict__ cnt,
                                float* __restrict__ pooled) {
  int i = blockIdx.x * blockDim.x + threadIdx.x;
  if (i >= GG * FD) return;
  pooled[i] = sums[i] / fmaxf(cnt[i >> 7], 1.0f);
}
__global__ void head1_kernel(const float* __restrict__ pooled, const float* __restrict__ W,
                             const float* __restrict__ b, float* __restrict__ t1) {
  int i = blockIdx.x * blockDim.x + threadIdx.x;   // GG*FD
  if (i >= GG * FD) return;
  int g = i >> 7, j = i & (FD - 1);
  float acc = b[j];
  for (int k = 0; k < FD; ++k) acc += pooled[g * FD + k] * W[k * FD + j];
  t1[i] = fmaxf(acc, 0.0f);
}
__global__ void head2_kernel(const float* __restrict__ t1, const float* __restrict__ W,
                             const float* __restrict__ b, float* __restrict__ out) {
  int i = blockIdx.x * blockDim.x + threadIdx.x;   // GG*CC
  if (i >= GG * CC) return;
  int g = i / CC, j = i % CC;
  float acc = b[j];
  for (int k = 0; k < FD; ++k) acc += t1[g * FD + k] * W[k * CC + j];
  out[i] = acc;
}

// ---------------------------------------------------------------- launcher
extern "C" void kernel_launch(void* const* d_in, const int* in_sizes, int n_in,
                              void* d_out, int out_size, void* d_ws, size_t ws_size,
                              hipStream_t stream) {
  const float* x     = (const float*)d_in[0];
  const int*   ei    = (const int*)d_in[1];
  const int*   batch = (const int*)d_in[2];
  const float* W1  = (const float*)d_in[3];
  const float* as1 = (const float*)d_in[4];
  const float* ad1 = (const float*)d_in[5];
  const float* b1  = (const float*)d_in[6];
  const float* W2  = (const float*)d_in[7];
  const float* as2 = (const float*)d_in[8];
  const float* ad2 = (const float*)d_in[9];
  const float* b2  = (const float*)d_in[10];
  const float* mW1 = (const float*)d_in[11];
  const float* mb1 = (const float*)d_in[12];
  const float* mW2 = (const float*)d_in[13];
  const float* mb2 = (const float*)d_in[14];
  const float* hW1 = (const float*)d_in[15];
  const float* hb1 = (const float*)d_in[16];
  const float* hW2 = (const float*)d_in[17];
  const float* hb2 = (const float*)d_in[18];

  // workspace carve-out (256B aligned)
  size_t off = 0;
  auto carve = [&](size_t bytes) -> char* {
    char* p = (char*)d_ws + off;
    off += (bytes + 255) & ~(size_t)255;
    return p;
  };
  float*     bufH  = (float*)carve((size_t)NN * FD * 4);  // GEMM output h
  float*     bufA  = (float*)carve((size_t)NN * FD * 4);  // layer-1 out / MLP hidden
  float*     bufB  = (float*)carve((size_t)NN * FD * 4);  // layer-2 out / node feats
  float*     hs    = (float*)carve((size_t)NN * 4);
  float*     hd    = (float*)carve((size_t)NN * 4);
  unsigned*  mord  = (unsigned*)carve((size_t)NN * 4);
  float*     ssum  = (float*)carve((size_t)NN * 4);
  float*     ew    = (float*)carve((size_t)ET * 4);
  _Float16*  Wt    = (_Float16*)carve((size_t)FD * FD * 2);
  float*     sums  = (float*)carve((size_t)GG * FD * 4);
  float*     cnt   = (float*)carve((size_t)GG * 4);
  float*     pooled= (float*)carve((size_t)GG * FD * 4);
  float*     t1    = (float*)carve((size_t)GG * FD * 4);

  const int T = 256;
  auto cdiv = [](long long a, long long b) { return (int)((a + b - 1) / b); };
  const int gemm_blocks  = (NN / 16) * (FD / 16) / (T / 32);  // 3125
  const int edge_blocks  = cdiv(ET, T);
  const int eagg_blocks  = cdiv((long long)ET * 32, T);
  const int elem_blocks  = cdiv((long long)NN * FD, T);
  const int node_blocks  = cdiv((long long)NN * 32, T);
  const int wtr_blocks   = cdiv(FD * FD, T);

  // one GAT layer: in -> out (out = relu(agg + bias))
  auto gat_layer = [&](const float* in, const float* W, const float* a_s, const float* a_d,
                       const float* bias, float* out) {
    wtrans_f16_kernel<<<wtr_blocks, T, 0, stream>>>(W, Wt);
    gemm_wmma_kernel<0><<<gemm_blocks, T, 0, stream>>>(in, Wt, nullptr, bufH, NN);
    node_scores_kernel<<<node_blocks, T, 0, stream>>>(bufH, a_s, a_d, hs, hd);
    fill_u32_kernel<<<cdiv(NN, T), T, 0, stream>>>(mord, ORD_NEG_INF, NN);
    fill_f32_kernel<<<cdiv(NN, T), T, 0, stream>>>(ssum, 0.0f, NN);
    fill_f32_kernel<<<elem_blocks, T, 0, stream>>>(out, 0.0f, NN * FD);
    edge_max_kernel<<<edge_blocks, T, 0, stream>>>(ei, hs, hd, ew, mord);
    edge_exp_kernel<<<edge_blocks, T, 0, stream>>>(ei, ew, mord, ssum);
    edge_agg_kernel<<<eagg_blocks, T, 0, stream>>>(ei, ew, ssum, bufH, out);
    bias_relu_kernel<<<elem_blocks, T, 0, stream>>>(out, bias);
  };

  // GAT layer 1: x -> bufA ; GAT layer 2: bufA -> bufB
  gat_layer(x, W1, as1, ad1, b1, bufA);
  gat_layer(bufA, W2, as2, ad2, b2, bufB);

  // node MLP: bufB -> bufA (relu) -> bufB
  wtrans_f16_kernel<<<wtr_blocks, T, 0, stream>>>(mW1, Wt);
  gemm_wmma_kernel<2><<<gemm_blocks, T, 0, stream>>>(bufB, Wt, mb1, bufA, NN);
  wtrans_f16_kernel<<<wtr_blocks, T, 0, stream>>>(mW2, Wt);
  gemm_wmma_kernel<1><<<gemm_blocks, T, 0, stream>>>(bufA, Wt, mb2, bufB, NN);

  // global mean pool
  fill_f32_kernel<<<cdiv(GG * FD, T), T, 0, stream>>>(sums, 0.0f, GG * FD);
  fill_f32_kernel<<<1, T, 0, stream>>>(cnt, 0.0f, GG);
  pool_sum_kernel<<<elem_blocks, T, 0, stream>>>(bufB, batch, sums, cnt);
  pool_div_kernel<<<cdiv(GG * FD, T), T, 0, stream>>>(sums, cnt, pooled);

  // head MLP (tiny, scalar)
  head1_kernel<<<cdiv(GG * FD, T), T, 0, stream>>>(pooled, hW1, hb1, t1);
  head2_kernel<<<1, 128, 0, stream>>>(t1, hW2, hb2, (float*)d_out);

  (void)in_sizes; (void)n_in; (void)out_size; (void)ws_size;
}
